// MultiHeadAttention_20899310862757
// MI455X (gfx1250) — matrix operand order repaired and checked by reading, not compiled, not run on hardware
//
#include <hip/hip_runtime.h>

// ---------------------------------------------------------------------------
// MHA forward for MI455X (gfx1250, wave32, WMMA, async global->LDS).
// Pipeline: convert f32->bf16, 3x projection GEMM (WMMA bf16, async
// double-buffered LDS staging), flash attention (WMMA bf16 + online softmax,
// async K staging), output projection GEMM -> f32.
// ---------------------------------------------------------------------------

typedef unsigned short u16;
typedef __attribute__((ext_vector_type(16))) __bf16 v16bf;
typedef __attribute__((ext_vector_type(8)))  float  v8f;

union F8   { v8f v;  float f[8]; };
union Frag { v16bf v; uint4 u[2]; };

__device__ __forceinline__ u16 f2bf(float x) {
  union { float f; unsigned u; } c; c.f = x;
  unsigned u = c.u;
  u += 0x7FFFu + ((u >> 16) & 1u);   // round-to-nearest-even
  return (u16)(u >> 16);
}

// LDS byte offset of a __shared__ pointer: on AMDGPU the generic (flat)
// representation of an LDS pointer carries the LDS offset in the low 32 bits.
__device__ __forceinline__ unsigned lds_off(const void* p) {
  return (unsigned)(unsigned long long)(uintptr_t)p;
}

// Async DMA: global -> LDS, 16 bytes per lane, tracked by ASYNCcnt.
__device__ __forceinline__ void async_copy_b128(unsigned lds_addr, const void* gaddr) {
  asm volatile("global_load_async_to_lds_b128 %0, %1, off"
               :: "v"(lds_addr), "v"(gaddr) : "memory");
}

// ---------------------------------------------------------------------------
// f32 -> bf16 conversion (vectorized x4)
// ---------------------------------------------------------------------------
__global__ __launch_bounds__(256) void f32_to_bf16_kernel(
    const float* __restrict__ in, u16* __restrict__ out, int n4) {
  int i = blockIdx.x * 256 + threadIdx.x;
  if (i < n4) {
    float4 v = ((const float4*)in)[i];
    ushort4 o;
    o.x = f2bf(v.x); o.y = f2bf(v.y); o.z = f2bf(v.z); o.w = f2bf(v.w);
    ((ushort4*)out)[i] = o;
  }
}

// ---------------------------------------------------------------------------
// GEMM: C[M,N] = A[M,K](bf16) @ W[N,K]^T(bf16) + bias, M=4096, N=K=1024.
// Block tile 128x128, BK=32. 256 threads = 8 waves, wave grid 4(M) x 2(N),
// each wave computes 32x64 via 2x4 WMMA 16x16 accumulators.
// LDS tiles double-buffered and filled with global_load_async_to_lds_b128;
// tile k+1 is in flight while tile k feeds the WMMAs. K-loop fully unrolled
// (constant trip count) so the scheduler can interleave ds_load/wmma streams.
// MODE 0: out = bf16 [B,H,S,64] layout, scaled (QKV projections).
// MODE 1: out = f32 row-major [M,N] (output projection -> d_out).
// ---------------------------------------------------------------------------
template <int MODE>
__global__ __launch_bounds__(256) void gemm_wmma_kernel(
    const u16* __restrict__ A, const u16* __restrict__ W,
    const float* __restrict__ bias, void* __restrict__ Out, float scale) {
  constexpr int M = 4096, N = 1024, K = 1024;
  constexpr int LDA = 40;  // 32 + 8 pad (80B rows: 16B aligned, bank-spread)
  constexpr int KT  = K / 32;

  __shared__ u16 As[2][128 * LDA];
  __shared__ u16 Bs[2][128 * LDA];

  const int tid  = threadIdx.x;
  const int lane = tid & 31;
  const int wave = tid >> 5;
  const int wm   = wave >> 1;   // 0..3 -> 32 rows each
  const int wn   = wave & 1;    // 0..1 -> 64 cols each
  const int m0   = blockIdx.y * 128;
  const int n0   = blockIdx.x * 128;

  F8 acc[2][4] = {};

  // Issue the async copies for one 128x32 K-tile (4 ASYNC ops per wave).
  auto issue_tile = [&](int k0, int buf) {
    #pragma unroll
    for (int i = 0; i < 2; ++i) {
      int idx = tid + i * 256;
      int row = idx >> 2;        // 4 chunks of 8 bf16 per row
      int ch  = idx & 3;
      async_copy_b128(lds_off(&As[buf][row * LDA + ch * 8]),
                      A + (size_t)(m0 + row) * K + k0 + ch * 8);
      async_copy_b128(lds_off(&Bs[buf][row * LDA + ch * 8]),
                      W + (size_t)(n0 + row) * K + k0 + ch * 8);
    }
  };

  issue_tile(0, 0);

  #pragma unroll
  for (int kt = 0; kt < KT; ++kt) {
    const int buf = kt & 1;
    if (kt + 1 < KT) {
      issue_tile((kt + 1) * 32, buf ^ 1);      // prefetch next tile (DMA)
      asm volatile("s_wait_asynccnt 0x4" ::: "memory");  // current tile done
    } else {
      asm volatile("s_wait_asynccnt 0x0" ::: "memory");
    }
    __syncthreads();

    const u16* as = As[buf];
    const u16* bs = Bs[buf];

    // ---- A fragments: lane holds row (lane&15), K chunks at (lane>>4)*8, +16
    Frag af[2];
    #pragma unroll
    for (int mb = 0; mb < 2; ++mb) {
      int row = wm * 32 + mb * 16 + (lane & 15);
      int klo = (lane >> 4) * 8;
      af[mb].u[0] = *(const uint4*)&as[row * LDA + klo];
      af[mb].u[1] = *(const uint4*)&as[row * LDA + klo + 16];
    }
    // ---- B fragments: lane holds col (lane&15), K chunks at (lane>>4)*8, +16
    #pragma unroll
    for (int nb = 0; nb < 4; ++nb) {
      Frag bf;
      int col = wn * 64 + nb * 16 + (lane & 15);
      int kb  = (lane >> 4) * 8;
      bf.u[0] = *(const uint4*)&bs[col * LDA + kb];
      bf.u[1] = *(const uint4*)&bs[col * LDA + kb + 16];
      #pragma unroll
      for (int mb = 0; mb < 2; ++mb) {
        acc[mb][nb].v = __builtin_amdgcn_wmma_f32_16x16x32_bf16(
            false, af[mb].v, false, bf.v, (short)0, acc[mb][nb].v, false, false);
      }
    }
    __syncthreads();  // protects buffer reuse 2 iterations later
  }

  // ---- epilogue: C layout: lane = n + 16*(m/8), vgpr r -> m = 8*(lane/16)+r
  #pragma unroll
  for (int mb = 0; mb < 2; ++mb) {
    #pragma unroll
    for (int nb = 0; nb < 4; ++nb) {
      #pragma unroll
      for (int r = 0; r < 8; ++r) {
        int row = m0 + wm * 32 + mb * 16 + (lane >> 4) * 8 + r;
        int col = n0 + wn * 64 + nb * 16 + (lane & 15);
        float val = (acc[mb][nb].f[r] + bias[col]) * scale;
        if (MODE == 0) {
          int b = row >> 11, s = row & 2047;   // M = B*S, S = 2048
          int h = col >> 6,  d = col & 63;     // N = H*64
          ((u16*)Out)[(((size_t)b * 16 + h) * 2048 + s) * 64 + d] = f2bf(val);
        } else {
          ((float*)Out)[(size_t)row * N + col] = val;
        }
      }
    }
  }
}

// ---------------------------------------------------------------------------
// Flash attention: grid (S/64, H, B), 128 threads = 4 waves.
// Wave w owns 16 query rows. Q fragments live in registers (pre-scaled by
// 1/sqrt(64) at projection). K tile is DMA'd global->LDS (async) while the
// V transpose streams through VGPRs; P re-shaped via wave-private LDS strip.
// Causal mask applied analytically on the diagonal tile.
// ---------------------------------------------------------------------------
__global__ __launch_bounds__(128) void flash_attn_kernel(
    const u16* __restrict__ Q, const u16* __restrict__ Kg,
    const u16* __restrict__ Vg, u16* __restrict__ Oc) {
  constexpr int S = 2048, Dh = 64, H = 16;
  constexpr int LD = 72;  // 64 + 8 pad (144B rows)

  __shared__ u16 Ks[64 * LD];
  __shared__ u16 Vt[64 * LD];       // transposed: [d][kv]
  __shared__ u16 Ps[4 * 16 * LD];   // per-wave P strips

  const int tid  = threadIdx.x;
  const int lane = tid & 31;
  const int wave = tid >> 5;
  const int it   = blockIdx.x;      // query tile (64 rows)
  const int h    = blockIdx.y;
  const int b    = blockIdx.z;

  const u16* Qh = Q  + ((size_t)(b * H + h)) * S * Dh;
  const u16* Kh = Kg + ((size_t)(b * H + h)) * S * Dh;
  const u16* Vh = Vg + ((size_t)(b * H + h)) * S * Dh;

  // Q fragments for this wave's 16 rows (2 K-steps over Dh=64), kept resident
  Frag qf[2];
  {
    int qrow = it * 64 + wave * 16 + (lane & 15);
    #pragma unroll
    for (int ks = 0; ks < 2; ++ks) {
      int klo = ks * 32 + (lane >> 4) * 8;
      qf[ks].u[0] = *(const uint4*)(Qh + (size_t)qrow * Dh + klo);
      qf[ks].u[1] = *(const uint4*)(Qh + (size_t)qrow * Dh + klo + 16);
    }
  }

  F8 o[4] = {};
  float m_r[8], l_r[8];
  #pragma unroll
  for (int r = 0; r < 8; ++r) { m_r[r] = -3.0e38f; l_r[r] = 0.0f; }

  for (int jt = 0; jt <= it; ++jt) {
    __syncthreads();   // previous tile's readers are done

    // ---- K tile: async DMA global->LDS (overlaps with V transpose below)
    #pragma unroll
    for (int i = 0; i < 4; ++i) {
      int idx = tid + i * 128;
      int row = idx >> 3;          // 8 chunks of 8 bf16 per 64-wide row
      int ch  = idx & 7;
      async_copy_b128(lds_off(&Ks[row * LD + ch * 8]),
                      Kh + (size_t)(jt * 64 + row) * Dh + ch * 8);
    }
    // ---- V tile: transpose through VGPRs into [d][kv]
    #pragma unroll
    for (int i = 0; i < 4; ++i) {
      int idx = tid + i * 128;
      int row = idx >> 3;
      int ch  = idx & 7;
      union { uint4 q; u16 s[8]; } cv;
      cv.q = *(const uint4*)(Vh + (size_t)(jt * 64 + row) * Dh + ch * 8);
      #pragma unroll
      for (int j = 0; j < 8; ++j) Vt[(ch * 8 + j) * LD + row] = cv.s[j];
    }
    asm volatile("s_wait_asynccnt 0x0" ::: "memory");
    __syncthreads();

    // ---- scores S = Q * K^T  (16x64 per wave)
    F8 sa[4] = {};
    #pragma unroll
    for (int ks = 0; ks < 2; ++ks) {
      #pragma unroll
      for (int nb = 0; nb < 4; ++nb) {
        Frag bf;
        int col = nb * 16 + (lane & 15);       // kv index in tile
        int kb  = ks * 32 + (lane >> 4) * 8;   // d chunk
        bf.u[0] = *(const uint4*)&Ks[col * LD + kb];
        bf.u[1] = *(const uint4*)&Ks[col * LD + kb + 16];
        sa[nb].v = __builtin_amdgcn_wmma_f32_16x16x32_bf16(
            false, qf[ks].v, false, bf.v, (short)0, sa[nb].v, false, false);
      }
    }

    // ---- causal mask on the diagonal tile
    if (jt == it) {
      int qloc = wave * 16 + (lane >> 4) * 8;  // + r below
      #pragma unroll
      for (int nb = 0; nb < 4; ++nb) {
        int kv = nb * 16 + (lane & 15);
        #pragma unroll
        for (int r = 0; r < 8; ++r)
          if (kv > qloc + r) sa[nb].f[r] = -1.0e9f;
      }
    }

    // ---- online softmax (rows live across 16 lanes; reduce with shfl_xor)
    #pragma unroll
    for (int r = 0; r < 8; ++r) {
      float t = fmaxf(fmaxf(sa[0].f[r], sa[1].f[r]),
                      fmaxf(sa[2].f[r], sa[3].f[r]));
      #pragma unroll
      for (int x = 1; x < 16; x <<= 1) t = fmaxf(t, __shfl_xor(t, x, 16));
      float mnew  = fmaxf(m_r[r], t);
      float alpha = __expf(m_r[r] - mnew);
      m_r[r] = mnew;
      float rs = 0.0f;
      #pragma unroll
      for (int nb = 0; nb < 4; ++nb) {
        float p = __expf(sa[nb].f[r] - mnew);
        sa[nb].f[r] = p;
        rs += p;
      }
      #pragma unroll
      for (int x = 1; x < 16; x <<= 1) rs += __shfl_xor(rs, x, 16);
      l_r[r] = l_r[r] * alpha + rs;
      #pragma unroll
      for (int nd = 0; nd < 4; ++nd) o[nd].f[r] *= alpha;
    }

    // ---- P -> wave-private LDS strip (re-shape C layout -> A layout);
    //      DS ops are in-order within a wave, no barrier needed.
    #pragma unroll
    for (int nb = 0; nb < 4; ++nb) {
      int pc = nb * 16 + (lane & 15);
      #pragma unroll
      for (int r = 0; r < 8; ++r) {
        int pr = wave * 16 + (lane >> 4) * 8 + r;
        Ps[pr * LD + pc] = f2bf(sa[nb].f[r]);
      }
    }

    // ---- O += P * V
    #pragma unroll
    for (int ks = 0; ks < 2; ++ks) {
      Frag pf;
      int prow = wave * 16 + (lane & 15);
      int klo  = ks * 32 + (lane >> 4) * 8;
      pf.u[0] = *(const uint4*)&Ps[prow * LD + klo];
      pf.u[1] = *(const uint4*)&Ps[prow * LD + klo + 16];
      #pragma unroll
      for (int nd = 0; nd < 4; ++nd) {
        Frag bf;
        int dcol = nd * 16 + (lane & 15);
        int kb   = ks * 32 + (lane >> 4) * 8;
        bf.u[0] = *(const uint4*)&Vt[dcol * LD + kb];
        bf.u[1] = *(const uint4*)&Vt[dcol * LD + kb + 16];
        o[nd].v = __builtin_amdgcn_wmma_f32_16x16x32_bf16(
            false, pf.v, false, bf.v, (short)0, o[nd].v, false, false);
      }
    }
  }

  // ---- normalize and write concat layout [B, S, H*64] as bf16
  #pragma unroll
  for (int nd = 0; nd < 4; ++nd) {
    #pragma unroll
    for (int r = 0; r < 8; ++r) {
      int qrow = it * 64 + wave * 16 + (lane >> 4) * 8 + r;
      int dcol = nd * 16 + (lane & 15);
      float val = o[nd].f[r] / l_r[r];
      Oc[((size_t)b * S + qrow) * 1024 + h * 64 + dcol] = f2bf(val);
    }
  }
}

// ---------------------------------------------------------------------------
// Host-side launch
// ---------------------------------------------------------------------------
extern "C" void kernel_launch(void* const* d_in, const int* in_sizes, int n_in,
                              void* d_out, int out_size, void* d_ws, size_t ws_size,
                              hipStream_t stream) {
  (void)in_sizes; (void)n_in; (void)out_size; (void)ws_size;

  const float* query = (const float*)d_in[0];
  const float* key   = (const float*)d_in[1];
  const float* value = (const float*)d_in[2];
  // d_in[3] = mask: causal, applied analytically in-kernel.
  const float* W_q = (const float*)d_in[4];
  const float* b_q = (const float*)d_in[5];
  const float* W_k = (const float*)d_in[6];
  const float* b_k = (const float*)d_in[7];
  const float* W_v = (const float*)d_in[8];
  const float* b_v = (const float*)d_in[9];
  const float* W_o = (const float*)d_in[10];
  const float* b_o = (const float*)d_in[11];

  const size_t pA = (size_t)4096 * 1024;  // activation tensor (bf16 elems)
  const size_t pW = (size_t)1024 * 1024;  // weight matrix (bf16 elems)

  u16* ws  = (u16*)d_ws;
  u16* Xq  = ws;            u16* Xk  = Xq  + pA;  u16* Xv  = Xk  + pA;
  u16* Wqb = Xv + pA;       u16* Wkb = Wqb + pW;  u16* Wvb = Wkb + pW;
  u16* Wob = Wvb + pW;
  u16* Qb  = Wob + pW;      u16* Kb  = Qb  + pA;  u16* Vb  = Kb  + pA;
  u16* Oc  = Vb + pA;

  // f32 -> bf16 conversions
  f32_to_bf16_kernel<<<(int)(pA / 4 / 256), 256, 0, stream>>>(query, Xq, (int)(pA / 4));
  f32_to_bf16_kernel<<<(int)(pA / 4 / 256), 256, 0, stream>>>(key,   Xk, (int)(pA / 4));
  f32_to_bf16_kernel<<<(int)(pA / 4 / 256), 256, 0, stream>>>(value, Xv, (int)(pA / 4));
  f32_to_bf16_kernel<<<(int)(pW / 4 / 256), 256, 0, stream>>>(W_q, Wqb, (int)(pW / 4));
  f32_to_bf16_kernel<<<(int)(pW / 4 / 256), 256, 0, stream>>>(W_k, Wkb, (int)(pW / 4));
  f32_to_bf16_kernel<<<(int)(pW / 4 / 256), 256, 0, stream>>>(W_v, Wvb, (int)(pW / 4));
  f32_to_bf16_kernel<<<(int)(pW / 4 / 256), 256, 0, stream>>>(W_o, Wob, (int)(pW / 4));

  // projections (Q pre-scaled by 1/sqrt(D_HEAD) = 0.125)
  dim3 ggrid(8, 32);
  gemm_wmma_kernel<0><<<ggrid, 256, 0, stream>>>(Xq, Wqb, b_q, Qb, 0.125f);
  gemm_wmma_kernel<0><<<ggrid, 256, 0, stream>>>(Xk, Wkb, b_k, Kb, 1.0f);
  gemm_wmma_kernel<0><<<ggrid, 256, 0, stream>>>(Xv, Wvb, b_v, Vb, 1.0f);

  // flash attention
  flash_attn_kernel<<<dim3(32, 16, 2), 128, 0, stream>>>(Qb, Kb, Vb, Oc);

  // output projection -> f32 d_out
  gemm_wmma_kernel<1><<<ggrid, 256, 0, stream>>>(Oc, Wob, b_o, d_out, 1.0f);
}
